// TransformerLayer_37022618092066
// MI455X (gfx1250) — compile-verified, matrix operand order checked by reading, not run on hardware
//
#include <hip/hip_runtime.h>

typedef __attribute__((ext_vector_type(16))) _Float16 v16h;
typedef __attribute__((ext_vector_type(8)))  _Float16 v8h;
typedef __attribute__((ext_vector_type(8)))  float    v8f;

#define LN_EPS 1e-5f
#define SLOPE  0.05f

// ---------------------------------------------------------------------------
// WMMA helpers (CDNA5 wave32, V_WMMA_F32_16X16X32_F16)
// 16-bit A 16x32 layout: lane = (r in 0..15 | half<<4); VGPRs 0..3 hold
// K = half*8 .. half*8+7 (consecutive), VGPRs 4..7 hold K = 16+half*8 .. +7.
// => one fragment = two 16-byte loads per lane from a row-major [rows][K] f16
// buffer. B operand uses the same loader on a pre-transposed [N][K] weight.
// ---------------------------------------------------------------------------
__device__ __forceinline__ v16h load_frag(const _Float16* base, int ld, int r0, int k0) {
  int lane = threadIdx.x & 31;
  int r  = lane & 15;
  int hh = lane >> 4;
  const _Float16* p = base + (long)(r0 + r) * ld + k0 + hh * 8;
  v8h lo = *(const v8h*)(p);        // K = k0 + hh*8 .. +7      -> VGPRs 0..3
  v8h hi = *(const v8h*)(p + 16);   // K = k0 + 16 + hh*8 .. +7 -> VGPRs 4..7
  return __builtin_shufflevector(lo, hi, 0,1,2,3,4,5,6,7,8,9,10,11,12,13,14,15);
}

__device__ __forceinline__ v8f wmma32(v16h a, v16h b, v8f c) {
  return __builtin_amdgcn_wmma_f32_16x16x32_f16(false, a, false, b, (short)0, c,
                                                false, false);
}

__device__ __forceinline__ v16h leaky_frag(v16h x) {
  v16h y;
#pragma unroll
  for (int i = 0; i < 16; ++i) {
    float f = (float)x[i];
    f = f >= 0.f ? f : SLOPE * f;
    y[i] = (_Float16)f;
  }
  return y;
}

// ---------------------------------------------------------------------------
// Prep kernels
// ---------------------------------------------------------------------------
__global__ void transpose_convert(const float* __restrict__ W, _Float16* __restrict__ Wt,
                                  int K, int N) {
  int total = K * N;
  for (int idx = blockIdx.x * blockDim.x + threadIdx.x; idx < total;
       idx += gridDim.x * blockDim.x) {
    int k = idx / N, n = idx % N;
    Wt[n * K + k] = (_Float16)W[idx];
  }
}

__global__ void convert_f16(const float* __restrict__ x, _Float16* __restrict__ y, int n) {
  for (int idx = blockIdx.x * blockDim.x + threadIdx.x; idx < n;
       idx += gridDim.x * blockDim.x)
    y[idx] = (_Float16)x[idx];
}

// ---------------------------------------------------------------------------
// Generic WMMA GEMM: C[M x N] = A_h[M x KK] @ W[KK x N] + bias  (W given as
// Wt[N x KK] f16).  8 waves; block tile 128x64; wave w owns rows w*16..+15.
// A fragments for the whole row-tile are hoisted into registers (KK/32 x v16h)
// and reused across all four 16-wide N sub-tiles.
// ---------------------------------------------------------------------------
template <int KK>
__global__ void __launch_bounds__(256)
gemm_wmma(const _Float16* __restrict__ A, const _Float16* __restrict__ Bt,
          const float* __restrict__ bias, float* __restrict__ Cf,
          _Float16* __restrict__ Ch, int M, int N, int act) {
  constexpr int NK = KK / 32;
  int w = threadIdx.x >> 5, lane = threadIdx.x & 31;
  int hh = lane >> 4, nl = lane & 15;
  int m0 = blockIdx.x * 128 + w * 16;

  v16h afr[NK];
#pragma unroll
  for (int kt = 0; kt < NK; ++kt) afr[kt] = load_frag(A, KK, m0, kt * 32);

  for (int nt = 0; nt < 4; ++nt) {
    int n0 = blockIdx.y * 64 + nt * 16;
    v8f acc = {};
#pragma unroll
    for (int kt = 0; kt < NK; ++kt) {
      v16h b = load_frag(Bt, KK, n0, kt * 32);
      acc = wmma32(afr[kt], b, acc);
    }
    int col = n0 + nl;
    float bv = bias ? bias[col] : 0.f;
#pragma unroll
    for (int r = 0; r < 8; ++r) {
      int row = m0 + r + 8 * hh;
      float v = acc[r] + bv;
      if (act) v = v >= 0.f ? v : SLOPE * v;
      if (Cf) Cf[(long)row * N + col] = v;
      if (Ch) Ch[(long)row * N + col] = (_Float16)v;
    }
  }
}

// ---------------------------------------------------------------------------
// Row LayerNorm over 256 dims (node path). One block (256 thr) per row.
// ---------------------------------------------------------------------------
__global__ void __launch_bounds__(256)
ln_rows_256(const float* __restrict__ x, const float* __restrict__ resid,
            const float* __restrict__ g, const float* __restrict__ bb, int leaky,
            float* __restrict__ outf, _Float16* __restrict__ outh) {
  __shared__ float red[8];
  int row = blockIdx.x, tid = threadIdx.x;
  float v = x[(long)row * 256 + tid] + resid[(long)row * 256 + tid];
  float s = v;
  for (int o = 16; o; o >>= 1) s += __shfl_xor(s, o, 32);
  if ((tid & 31) == 0) red[tid >> 5] = s;
  __syncthreads();
  float tot = 0.f;
  for (int k = 0; k < 8; ++k) tot += red[k];
  float m = tot * (1.f / 256.f);
  float d = v - m;
  float s2 = d * d;
  for (int o = 16; o; o >>= 1) s2 += __shfl_xor(s2, o, 32);
  __syncthreads();
  if ((tid & 31) == 0) red[tid >> 5] = s2;
  __syncthreads();
  float tv = 0.f;
  for (int k = 0; k < 8; ++k) tv += red[k];
  float rstd = rsqrtf(tv * (1.f / 256.f) + LN_EPS);
  float y = d * rstd * g[tid] + bb[tid];
  if (leaky) y = y >= 0.f ? y : SLOPE * y;
  if (outf) outf[(long)row * 256 + tid] = y;
  if (outh) outh[(long)row * 256 + tid] = (_Float16)y;
}

// ---------------------------------------------------------------------------
// Fused edge-path kernel: one workgroup per (b, i).  Everything between
// reading edges[b,i,:,:] (once) and writing new_edges[b,i,:,:] stays in LDS.
//   em/ea GEMMs -> new_e (f16, LDS) -> logits -> softmax -> wv
//   -> leaky(new_e) @ Weo + resid -> LN -> edge MLP (2 GEMMs) -> LN -> store
// LDS budget: 152 KB dynamic (2 workgroups per 320 KB WGP).
// The f32 edge slice (residual source) time-aliases the sAE/sH1 buffers.
// ---------------------------------------------------------------------------
__global__ void __launch_bounds__(256)
edge_fused(const float* __restrict__ edges, const float* __restrict__ qf,
           const float* __restrict__ kf, const float* __restrict__ vf,
           const _Float16* __restrict__ WmulT, const float* __restrict__ bmul,
           const _Float16* __restrict__ WaddT, const float* __restrict__ badd,
           const _Float16* __restrict__ WeoT,  const float* __restrict__ beo,
           const float* __restrict__ g1e, const float* __restrict__ b1e,
           const _Float16* __restrict__ We1T, const float* __restrict__ be1,
           const _Float16* __restrict__ We2T, const float* __restrict__ be2,
           const float* __restrict__ g2e, const float* __restrict__ b2e,
           float* __restrict__ wv_out, float* __restrict__ edges_out) {
  extern __shared__ char smem[];
  _Float16* sNewE = (_Float16*)smem;              // [128][256] f16  (64 KB)
  float*    sT    = (float*)smem;                 // alias: [128][64] f32 (32 KB)
  float*    sAEf  = (float*)(smem + 65536);       // [128][64] f32   (32 KB)
  _Float16* sE    = (_Float16*)(smem + 98304);    // [128][64] f16   (16 KB)
  float*    sEf   = (float*)(smem + 114688);      // [128][64] f32   (32 KB)
  _Float16* sAE   = (_Float16*)(smem + 114688);   // aliases sEf lo  (16 KB)
  _Float16* sH1   = (_Float16*)(smem + 131072);   // aliases sEf hi  (16 KB)
  float*    sLog  = (float*)(smem + 147456);      // [128][8]        (4 KB)
  float*    sProb = (float*)(smem + 151552);      // [128][8]        (4 KB)

  int tid = threadIdx.x;
  int w = tid >> 5, lane = tid & 31, hh = lane >> 4, nl = lane & 15;
  int blk = blockIdx.x;
  int b = blk >> 7, i = blk & 127;
  int j0 = w * 16;  // row sub-tile owned by this wave

  const float* erow  = edges + (long)(b * 128 + i) * 128 * 64;  // [128][64]
  const float* qrow  = qf + (long)(b * 128 + i) * 256;
  const float* kbase = kf + (long)b * 128 * 256;
  const float* vbase = vf + (long)b * 128 * 256;

  // 0) warm the shared weight tiles (reused by all 1024 blocks) into cache
  //    -> lowers to global_prefetch_b8
  {
    int off = tid * 128;                 // 256 thr x 128 B = 32 KB
    __builtin_prefetch((const char*)WmulT + off, 0, 1);
    __builtin_prefetch((const char*)WaddT + off, 0, 1);
    __builtin_prefetch((const char*)WeoT + off, 0, 1);
  }

  // 1) stage edge slice once: f32 copy (residual) + f16 copy (GEMM A operand)
  for (int idx = tid; idx < 128 * 64; idx += 256) {
    float e = erow[idx];
    sEf[idx] = e;
    sE[idx] = (_Float16)e;
  }
  __syncthreads();

  // 2) em/ea = E @ Wmul/Wadd  (M=128,K=64,N=256), fused elementwise new_e
  {
    v16h aE[2];
    aE[0] = load_frag(sE, 64, j0, 0);
    aE[1] = load_frag(sE, 64, j0, 32);
    for (int nt = 0; nt < 16; ++nt) {
      int n0 = nt * 16;
      v8f am = {}, aa = {};
#pragma unroll
      for (int kt = 0; kt < 2; ++kt) {
        v16h bm = load_frag(WmulT, 64, n0, kt * 32);
        am = wmma32(aE[kt], bm, am);
        v16h ba = load_frag(WaddT, 64, n0, kt * 32);
        aa = wmma32(aE[kt], ba, aa);
      }
      int d = n0 + nl;
      float qd = qrow[d] * 0.0625f;  // / sqrt(256)
      float bm_ = bmul[d], ba_ = badd[d];
#pragma unroll
      for (int r = 0; r < 8; ++r) {
        int j = j0 + r + 8 * hh;
        float attn = qd * kbase[j * 256 + d];
        float ne = attn * (am[r] + bm_) + attn + (aa[r] + ba_);
        sNewE[j * 256 + d] = (_Float16)ne;
      }
    }
  }
  __syncthreads();

  // 3) logits[j][h] = sum_d new_e
  for (int idx = tid; idx < 1024; idx += 256) {
    int j = idx >> 3, h = idx & 7;
    const _Float16* p = sNewE + j * 256 + h * 32;
    float s = 0.f;
    for (int d = 0; d < 32; ++d) s += (float)p[d];
    sLog[idx] = s;
  }
  __syncthreads();

  // 4) softmax over j (128 keys) per head; wave w handles head w
  {
    int h = w;
    float l0 = sLog[(lane)      * 8 + h];
    float l1 = sLog[(lane + 32) * 8 + h];
    float l2 = sLog[(lane + 64) * 8 + h];
    float l3 = sLog[(lane + 96) * 8 + h];
    float m = fmaxf(fmaxf(l0, l1), fmaxf(l2, l3));
    for (int o = 16; o; o >>= 1) m = fmaxf(m, __shfl_xor(m, o, 32));
    float e0 = expf(l0 - m), e1 = expf(l1 - m), e2 = expf(l2 - m), e3 = expf(l3 - m);
    float s = e0 + e1 + e2 + e3;
    for (int o = 16; o; o >>= 1) s += __shfl_xor(s, o, 32);
    float inv = 1.f / s;
    sProb[(lane)      * 8 + h] = e0 * inv;
    sProb[(lane + 32) * 8 + h] = e1 * inv;
    sProb[(lane + 64) * 8 + h] = e2 * inv;
    sProb[(lane + 96) * 8 + h] = e3 * inv;
  }
  __syncthreads();

  // 5) wv[i, t] = sum_j a[j, t>>5] * v[b, j, t]  (t = h*32 + d)
  {
    int h = tid >> 5;
    float acc = 0.f;
    for (int j = 0; j < 128; ++j) acc += sProb[j * 8 + h] * vbase[j * 256 + tid];
    wv_out[(long)(b * 128 + i) * 256 + tid] = acc;
  }

  // 6) attn_edges = leaky(new_e)[128x256] @ Weo + beo + edges  (K=256,N=64)
  {
    v16h aN[8];
#pragma unroll
    for (int kt = 0; kt < 8; ++kt)
      aN[kt] = leaky_frag(load_frag(sNewE, 256, j0, kt * 32));
    for (int nt = 0; nt < 4; ++nt) {
      int n0 = nt * 16;
      v8f acc = {};
#pragma unroll
      for (int kt = 0; kt < 8; ++kt) {
        v16h bb = load_frag(WeoT, 256, n0, kt * 32);
        acc = wmma32(aN[kt], bb, acc);
      }
      int n = n0 + nl;
      float bv = beo[n];
#pragma unroll
      for (int r = 0; r < 8; ++r) {
        int j = j0 + r + 8 * hh;
        sAEf[j * 64 + n] = acc[r] + bv + sEf[j * 64 + n];
      }
    }
  }
  __syncthreads();

  // 7) LN over 64 dims per row (g1e/b1e); keep f32 (residual) + f16 (GEMM A).
  //    sAE overwrites the (now dead) sEf region -- barrier-separated.
  if (tid < 128) {
    int j = tid;
    float* row = sAEf + j * 64;
    float s = 0.f;
    for (int n = 0; n < 64; ++n) s += row[n];
    float m = s * (1.f / 64.f);
    float v2 = 0.f;
    for (int n = 0; n < 64; ++n) { float dd = row[n] - m; v2 += dd * dd; }
    float rstd = rsqrtf(v2 * (1.f / 64.f) + LN_EPS);
    for (int n = 0; n < 64; ++n) {
      float y = (row[n] - m) * rstd * g1e[n] + b1e[n];
      row[n] = y;
      sAE[j * 64 + n] = (_Float16)y;
    }
  }
  __syncthreads();

  // 8) h1 = leaky(attn_edges @ We1 + be1)   (64x64)
  {
    v16h aA[2];
    aA[0] = load_frag(sAE, 64, j0, 0);
    aA[1] = load_frag(sAE, 64, j0, 32);
    for (int nt = 0; nt < 4; ++nt) {
      int n0 = nt * 16;
      v8f acc = {};
#pragma unroll
      for (int kt = 0; kt < 2; ++kt) {
        v16h bb = load_frag(We1T, 64, n0, kt * 32);
        acc = wmma32(aA[kt], bb, acc);
      }
      int n = n0 + nl;
      float bv = be1[n];
#pragma unroll
      for (int r = 0; r < 8; ++r) {
        int j = j0 + r + 8 * hh;
        float v = acc[r] + bv;
        sH1[j * 64 + n] = (_Float16)(v >= 0.f ? v : SLOPE * v);
      }
    }
  }
  __syncthreads();

  // 9) ne2 = h1 @ We2 + be2; pre-LN residual into sT (aliases dead sNewE)
  {
    v16h aH[2];
    aH[0] = load_frag(sH1, 64, j0, 0);
    aH[1] = load_frag(sH1, 64, j0, 32);
    for (int nt = 0; nt < 4; ++nt) {
      int n0 = nt * 16;
      v8f acc = {};
#pragma unroll
      for (int kt = 0; kt < 2; ++kt) {
        v16h bb = load_frag(We2T, 64, n0, kt * 32);
        acc = wmma32(aH[kt], bb, acc);
      }
      int n = n0 + nl;
      float bv = be2[n];
#pragma unroll
      for (int r = 0; r < 8; ++r) {
        int j = j0 + r + 8 * hh;
        sT[j * 64 + n] = acc[r] + bv + sAEf[j * 64 + n];
      }
    }
  }
  __syncthreads();

  // 10) final LN (g2e/b2e), write new_edges[b,i,:,:]
  if (tid < 128) {
    int j = tid;
    float* row = sT + j * 64;
    float s = 0.f;
    for (int n = 0; n < 64; ++n) s += row[n];
    float m = s * (1.f / 64.f);
    float v2 = 0.f;
    for (int n = 0; n < 64; ++n) { float dd = row[n] - m; v2 += dd * dd; }
    float rstd = rsqrtf(v2 * (1.f / 64.f) + LN_EPS);
    float* orow = edges_out + ((long)(b * 128 + i) * 128 + j) * 64;
    for (int n = 0; n < 64; ++n)
      orow[n] = (row[n] - m) * rstd * g2e[n] + b2e[n];
  }
}

// ---------------------------------------------------------------------------
// Host-side orchestration
// ---------------------------------------------------------------------------
extern "C" void kernel_launch(void* const* d_in, const int* in_sizes, int n_in,
                              void* d_out, int out_size, void* d_ws, size_t ws_size,
                              hipStream_t stream) {
  (void)in_sizes; (void)n_in; (void)out_size; (void)ws_size;

  const float* nodes = (const float*)d_in[0];
  const float* edges = (const float*)d_in[1];
  const float *Wq = (const float*)d_in[2],  *bq  = (const float*)d_in[3];
  const float *Wk = (const float*)d_in[4],  *bk  = (const float*)d_in[5];
  const float *Wv = (const float*)d_in[6],  *bv  = (const float*)d_in[7];
  const float *Wmul = (const float*)d_in[8],  *bmul = (const float*)d_in[9];
  const float *Wadd = (const float*)d_in[10], *badd = (const float*)d_in[11];
  const float *Wno = (const float*)d_in[12], *bno = (const float*)d_in[13];
  const float *Weo = (const float*)d_in[14], *beo = (const float*)d_in[15];
  const float *g1n = (const float*)d_in[16], *b1n = (const float*)d_in[17];
  const float *g1e = (const float*)d_in[18], *b1e = (const float*)d_in[19];
  const float *Wn1 = (const float*)d_in[20], *bn1 = (const float*)d_in[21];
  const float *Wn2 = (const float*)d_in[22], *bn2 = (const float*)d_in[23];
  const float *We1 = (const float*)d_in[24], *be1 = (const float*)d_in[25];
  const float *We2 = (const float*)d_in[26], *be2 = (const float*)d_in[27];
  const float *g2n = (const float*)d_in[28], *b2n = (const float*)d_in[29];
  const float *g2e = (const float*)d_in[30], *b2e = (const float*)d_in[31];

  // workspace carve-out (256-B aligned)
  char* p = (char*)d_ws;
  auto carve = [&](size_t bytes) -> void* {
    void* r = (void*)p;
    p += (bytes + 255) & ~(size_t)255;
    return r;
  };
  _Float16* WqT  = (_Float16*)carve(256 * 256 * 2);
  _Float16* WkT  = (_Float16*)carve(256 * 256 * 2);
  _Float16* WvT  = (_Float16*)carve(256 * 256 * 2);
  _Float16* WnoT = (_Float16*)carve(256 * 256 * 2);
  _Float16* Wn1T = (_Float16*)carve(256 * 256 * 2);
  _Float16* Wn2T = (_Float16*)carve(256 * 256 * 2);
  _Float16* WmulT = (_Float16*)carve(256 * 64 * 2);
  _Float16* WaddT = (_Float16*)carve(256 * 64 * 2);
  _Float16* WeoT  = (_Float16*)carve(64 * 256 * 2);
  _Float16* We1T  = (_Float16*)carve(64 * 64 * 2);
  _Float16* We2T  = (_Float16*)carve(64 * 64 * 2);
  const int NNODE = 8 * 128 * 256;  // 262144
  _Float16* nodes_h = (_Float16*)carve(NNODE * 2);
  float* q_f   = (float*)carve(NNODE * 4);
  float* k_f   = (float*)carve(NNODE * 4);
  float* v_f   = (float*)carve(NNODE * 4);
  float* wv_f  = (float*)carve(NNODE * 4);
  float* tmp_f = (float*)carve(NNODE * 4);
  float* an1_f = (float*)carve(NNODE * 4);
  float* nn2_f = (float*)carve(NNODE * 4);
  _Float16* wv_h  = (_Float16*)carve(NNODE * 2);
  _Float16* an1_h = (_Float16*)carve(NNODE * 2);
  _Float16* h1_h  = (_Float16*)carve(NNODE * 2);

  float* out_nodes = (float*)d_out;
  float* out_edges = (float*)d_out + NNODE;

  // 1) weight prep (f16, transposed to [N][K])
  transpose_convert<<<dim3(256), dim3(256), 0, stream>>>(Wq,  WqT,  256, 256);
  transpose_convert<<<dim3(256), dim3(256), 0, stream>>>(Wk,  WkT,  256, 256);
  transpose_convert<<<dim3(256), dim3(256), 0, stream>>>(Wv,  WvT,  256, 256);
  transpose_convert<<<dim3(256), dim3(256), 0, stream>>>(Wno, WnoT, 256, 256);
  transpose_convert<<<dim3(256), dim3(256), 0, stream>>>(Wn1, Wn1T, 256, 256);
  transpose_convert<<<dim3(256), dim3(256), 0, stream>>>(Wn2, Wn2T, 256, 256);
  transpose_convert<<<dim3(64),  dim3(256), 0, stream>>>(Wmul, WmulT, 64, 256);
  transpose_convert<<<dim3(64),  dim3(256), 0, stream>>>(Wadd, WaddT, 64, 256);
  transpose_convert<<<dim3(64),  dim3(256), 0, stream>>>(Weo,  WeoT,  256, 64);
  transpose_convert<<<dim3(16),  dim3(256), 0, stream>>>(We1,  We1T,  64, 64);
  transpose_convert<<<dim3(16),  dim3(256), 0, stream>>>(We2,  We2T,  64, 64);
  convert_f16<<<dim3(512), dim3(256), 0, stream>>>(nodes, nodes_h, NNODE);

  // 2) q/k/v projections  (M=1024, N=256, K=256)
  gemm_wmma<256><<<dim3(8, 4), dim3(256), 0, stream>>>(nodes_h, WqT, bq, q_f,
                                                       nullptr, 1024, 256, 0);
  gemm_wmma<256><<<dim3(8, 4), dim3(256), 0, stream>>>(nodes_h, WkT, bk, k_f,
                                                       nullptr, 1024, 256, 0);
  gemm_wmma<256><<<dim3(8, 4), dim3(256), 0, stream>>>(nodes_h, WvT, bv, v_f,
                                                       nullptr, 1024, 256, 0);

  // 3) fused edge path; one workgroup per (b,i); 152 KB dynamic LDS
  edge_fused<<<dim3(1024), dim3(256), 155648, stream>>>(
      edges, q_f, k_f, v_f, WmulT, bmul, WaddT, badd, WeoT, beo, g1e, b1e,
      We1T, be1, We2T, be2, g2e, b2e, wv_f, out_edges);

  // 4) node tail
  convert_f16<<<dim3(512), dim3(256), 0, stream>>>(wv_f, wv_h, NNODE);
  gemm_wmma<256><<<dim3(8, 4), dim3(256), 0, stream>>>(wv_h, WnoT, bno, tmp_f,
                                                       nullptr, 1024, 256, 0);
  ln_rows_256<<<dim3(1024), dim3(256), 0, stream>>>(tmp_f, nodes, g1n, b1n, 1,
                                                    an1_f, an1_h);
  gemm_wmma<256><<<dim3(8, 4), dim3(256), 0, stream>>>(an1_h, Wn1T, bn1, nullptr,
                                                       h1_h, 1024, 256, 1);
  gemm_wmma<256><<<dim3(8, 4), dim3(256), 0, stream>>>(h1_h, Wn2T, bn2, nn2_f,
                                                       nullptr, 1024, 256, 0);
  ln_rows_256<<<dim3(1024), dim3(256), 0, stream>>>(nn2_f, an1_f, g2n, b2n, 1,
                                                    out_nodes, nullptr);
}
